// AblationGNNTracker_36550171689331
// MI455X (gfx1250) — matrix operand order repaired and checked by reading, compiled, not run on hardware
//
#include <hip/hip_runtime.h>
#include <hip/hip_bf16.h>

typedef __bf16 bf16_t;
typedef __bf16 v8bf  __attribute__((ext_vector_type(8)));
typedef __bf16 v16bf __attribute__((ext_vector_type(16)));
typedef float  v8f   __attribute__((ext_vector_type(8)));
typedef unsigned v4u __attribute__((ext_vector_type(4)));
typedef int      v4i __attribute__((ext_vector_type(4)));
typedef int      v8i __attribute__((ext_vector_type(8)));

#define GF_ACCUM   1
#define GF_GELU    2
#define GF_STOREBF 4

#if defined(__has_builtin)
#if __has_builtin(__builtin_amdgcn_tensor_load_to_lds) && __has_builtin(__builtin_amdgcn_s_wait_tensorcnt)
#define HAVE_TDM 1
#endif
#endif
#ifndef HAVE_TDM
#define HAVE_TDM 0
#endif

__device__ __forceinline__ float gelu_f(float x) {
  return 0.5f * x * (1.0f + erff(x * 0.70710678118654752f));
}
__device__ __forceinline__ float softplus_f(float x) {
  return x > 20.0f ? x : log1pf(expf(x));
}

// ---------------- weight pack: f32 row-major (Krows, Ncols) -> bf16 transposed (Ncols, Kpad), zero padded
__global__ void k_pack(const float* __restrict__ W, bf16_t* __restrict__ Wt,
                       int Krows, int Ncols, int Kpad) {
  int i = blockIdx.x * blockDim.x + threadIdx.x;
  if (i >= Ncols * Kpad) return;
  int col = i / Kpad, kp = i % Kpad;
  Wt[(long)col * Kpad + kp] = (kp < Krows) ? (bf16_t)W[(long)kp * Ncols + col] : (bf16_t)0.0f;
}

// ---------------- x_in = [x, sin(2pi x@B), cos(2pi x@B), pad] as bf16, stride 96
__global__ void k_build_xin(const float* __restrict__ x, const float* __restrict__ B,
                            bf16_t* __restrict__ xin, int N) {
  int i = blockIdx.x * blockDim.x + threadIdx.x;
  if (i >= N * 32) return;
  int n = i >> 5, j = i & 31;
  float x0 = x[n * 2], x1 = x[n * 2 + 1];
  float xp = 6.28318530717958647f * (x0 * B[j] + x1 * B[32 + j]);
  bf16_t* row = xin + (long)n * 96;
  row[2 + j]  = (bf16_t)sinf(xp);
  row[34 + j] = (bf16_t)cosf(xp);
  if (j < 2)  row[j] = (bf16_t)(j == 0 ? x0 : x1);
  if (j < 30) row[66 + j] = (bf16_t)0.0f;
}

// ---------------- edge_attr (E,8) -> bf16 padded to stride 32
__global__ void k_build_ein(const float* __restrict__ ea, bf16_t* __restrict__ ein, int E) {
  int i = blockIdx.x * blockDim.x + threadIdx.x;
  if (i >= E * 32) return;
  int e = i >> 5, j = i & 31;
  ein[(long)e * 32 + j] = (j < 8) ? (bf16_t)ea[(long)e * 8 + j] : (bf16_t)0.0f;
}

// ---------------- WMMA GEMM: Y[M,128] (+=) X[idx?,KP]bf16 @ Wt[128,KP]bf16 + bias
// Weight tile staged in LDS (TDM async-tensor load when available), 16B row pad for
// bank-conflict-free ds_load_b128 B-fragment reads. 8 waves x (16 rows x 128 cols).
template <int KP>
__global__ __launch_bounds__(256) void k_gemm(
    const bf16_t* __restrict__ X, int ldx, const int* __restrict__ idx,
    const bf16_t* __restrict__ Wt, const float* __restrict__ bias,
    float* __restrict__ Y, bf16_t* __restrict__ Ybf,
    int M, int flags) {
  constexpr bool PAD = (KP == 128 || KP == 32);   // row dwords power-of-2 -> TDM pad expressible
  constexpr int  KS  = PAD ? KP + 8 : KP;         // LDS row stride (bf16 elems), +16B pad
  __shared__ bf16_t sW[128 * KS];

#if HAVE_TDM
  if (threadIdx.x < 32) {  // wave 0 issues one TDM descriptor (EXEC ignored by tensor ops)
    unsigned lds_off = (unsigned)(unsigned long long)(void*)&sW[0];  // addr[31:0] = LDS offset
    unsigned long long ga = (unsigned long long)Wt;
    const unsigned Q = (unsigned)(128u * (unsigned)KP * 2u / 8u);    // qwords in tile
    constexpr unsigned PADEN = PAD ? 1u : 0u;
    constexpr unsigned PI = (KP == 128) ? 5u : 3u;  // pad interval = 2^PI qwords (one row)
    constexpr unsigned PA = 3u;                     // pad amount = 4 dwords = 16B
    v4u g0;
    g0[0] = 1u;                                        // count=1, user descriptor
    g0[1] = lds_off;                                   // lds_addr
    g0[2] = (unsigned)(ga & 0xFFFFFFFFu);              // global_addr[31:0]
    g0[3] = (unsigned)((ga >> 32) & 0x01FFFFFFull) | 0x80000000u;  // ga[56:32] | type=2
    v8i g1;
    g1[0] = (int)((3u << 16) | (PADEN << 20) | (PADEN ? ((PI << 22) | (PA << 25)) : 0u)); // data_size=8B
    g1[1] = (int)((Q & 0xFFFFu) << 16);                // tensor_dim0[15:0]
    g1[2] = (int)(((Q >> 16) & 0xFFFFu) | (1u << 16)); // tensor_dim0[31:16] | tensor_dim1=1
    g1[3] = (int)((Q & 0xFFFFu) << 16);                // tile_dim0 = Q (<=4096, fits 16b)
    g1[4] = 0;                                         // tile_dim1=0 (1D), tile_dim2=0
    g1[5] = (int)Q;                                    // tensor_dim0_stride lo32
    g1[6] = 0;
    g1[7] = 0;
    v4i gz4 = {0, 0, 0, 0};
    v8i gz8 = {0, 0, 0, 0, 0, 0, 0, 0};
    __builtin_amdgcn_tensor_load_to_lds(g0, g1, gz4, gz4, gz8, 0);
    __builtin_amdgcn_s_wait_tensorcnt(0);
  }
#else
  {
    constexpr int ROWCH = KP / 8;        // 16B chunks per row
    for (int c = threadIdx.x; c < 16 * KP; c += 256) {
      int row = c / ROWCH, w = c % ROWCH;
      *(v8bf*)(sW + (long)row * KS + w * 8) = *(const v8bf*)(Wt + (long)row * KP + w * 8);
    }
  }
#endif
  __syncthreads();

  const int lane = threadIdx.x & 31;
  const int wave = threadIdx.x >> 5;
  const int rbase = blockIdx.x * 128 + wave * 16;
  if (rbase >= M) return;  // wave-uniform, after the barrier

  int arow = rbase + (lane & 15);
  if (arow >= M) arow = M - 1;                 // clamp; rows >= M are never stored
  long xrow = idx ? (long)idx[arow] : (long)arow;
  const bf16_t* aptr = X + xrow * (long)ldx + (lane >> 4) * 8;

  v8f zero = {0.f, 0.f, 0.f, 0.f, 0.f, 0.f, 0.f, 0.f};
  v8f acc[8];
#pragma unroll
  for (int t = 0; t < 8; ++t) acc[t] = zero;

  const int bk = (lane >> 4) * 8;
  const int bcol = lane & 15;

#pragma unroll
  for (int kk = 0; kk < KP; kk += 32) {
    union { v16bf v; v8bf h[2]; } A;
    A.h[0] = *(const v8bf*)(aptr + kk);
    A.h[1] = *(const v8bf*)(aptr + kk + 16);
    union { v16bf v; v8bf h[2]; } Bm[8];
#pragma unroll
    for (int t = 0; t < 8; ++t) {                 // batch all B-frag LDS loads
      const bf16_t* bp = sW + (t * 16 + bcol) * KS + kk + bk;
      Bm[t].h[0] = *(const v8bf*)(bp);
      Bm[t].h[1] = *(const v8bf*)(bp + 16);
    }
#pragma unroll
    for (int t = 0; t < 8; ++t)                   // back-to-back WMMAs, no per-op waits
      acc[t] = __builtin_amdgcn_wmma_f32_16x16x32_bf16(
          false, A.v, false, Bm[t].v, (short)0, acc[t], false, false);
  }

  const int rrow0 = rbase + (lane >> 4) * 8;
#pragma unroll
  for (int t = 0; t < 8; ++t) {
    int col = t * 16 + (lane & 15);
    float bv = bias ? bias[col] : 0.0f;
#pragma unroll
    for (int j = 0; j < 8; ++j) {
      int r = rrow0 + j;
      if (r < M) {
        long o = (long)r * 128 + col;
        float v = acc[t][j] + bv;
        if (flags & GF_ACCUM) v += Y[o];
        if (flags & GF_GELU)  v = gelu_f(v);
        Y[o] = v;
        if (flags & GF_STOREBF) Ybf[o] = (bf16_t)v;
      }
    }
  }
}

// ---------------- LayerNorm(g,beta) + GELU -> bf16, one row per block
__global__ __launch_bounds__(128) void k_ln_act(const float* __restrict__ Y,
                                                const float* __restrict__ g,
                                                const float* __restrict__ b,
                                                bf16_t* __restrict__ obf, int M) {
  __shared__ float red[128];
  int row = blockIdx.x;
  int c = threadIdx.x;
  float v = Y[(long)row * 128 + c];
  red[c] = v; __syncthreads();
  for (int s = 64; s > 0; s >>= 1) { if (c < s) red[c] += red[c + s]; __syncthreads(); }
  float mu = red[0] * (1.0f / 128.0f);
  __syncthreads();
  float d = v - mu;
  red[c] = d * d; __syncthreads();
  for (int s = 64; s > 0; s >>= 1) { if (c < s) red[c] += red[c + s]; __syncthreads(); }
  float var = red[0] * (1.0f / 128.0f);
  float xn = d * rsqrtf(var + 1e-5f) * g[c] + b[c];
  obf[(long)row * 128 + c] = (bf16_t)gelu_f(xn);
}

// ---------------- attention: score + segment max (order-preserving uint atomicMax)
__global__ __launch_bounds__(256) void k_score_max(
    const float* __restrict__ q, const float* __restrict__ k, const float* __restrict__ eh,
    const int* __restrict__ src, const int* __restrict__ dst,
    float* __restrict__ score, unsigned* __restrict__ mkey, int E) {
  int w = blockIdx.x * 8 + (threadIdx.x >> 5);
  if (w >= E * 4) return;
  int lane = threadIdx.x & 31;
  int e = w >> 2, hh = w & 3;
  int s_ = src[e], d_ = dst[e];
  int f = hh * 32 + lane;
  float qv = q[(long)d_ * 128 + f];
  float kv = k[(long)s_ * 128 + f] + eh[(long)e * 128 + f];
  float p = qv * kv;
  for (int off = 16; off; off >>= 1) p += __shfl_xor(p, off, 32);
  if (lane == 0) {
    float sc = p * 0.17677669529663689f;  // 1/sqrt(32)
    score[(long)e * 4 + hh] = sc;
    unsigned u = __float_as_uint(sc);
    u = (u & 0x80000000u) ? ~u : (u | 0x80000000u);
    atomicMax(&mkey[d_ * 4 + hh], u);
  }
}

__global__ void k_exp_denom(float* __restrict__ score, const unsigned* __restrict__ mkey,
                            float* __restrict__ denom, const int* __restrict__ dst, int EH) {
  int i = blockIdx.x * blockDim.x + threadIdx.x;
  if (i >= EH) return;
  int e = i >> 2, hh = i & 3;
  int d_ = dst[e];
  unsigned u = mkey[d_ * 4 + hh];
  float m = (u & 0x80000000u) ? __uint_as_float(u & 0x7FFFFFFFu) : __uint_as_float(~u);
  float a = expf(score[i] - m);
  score[i] = a;
  atomicAdd(&denom[d_ * 4 + hh], a);
}

__global__ __launch_bounds__(256) void k_msg(
    const float* __restrict__ v, const float* __restrict__ eh,
    const float* __restrict__ score, const float* __restrict__ denom,
    const int* __restrict__ src, const int* __restrict__ dst,
    float* __restrict__ outb, int E) {
  int w = blockIdx.x * 8 + (threadIdx.x >> 5);
  if (w >= E * 4) return;
  int lane = threadIdx.x & 31;
  int e = w >> 2, hh = w & 3;
  int s_ = src[e], d_ = dst[e];
  float alpha = score[(long)e * 4 + hh] / denom[d_ * 4 + hh];
  int f = hh * 32 + lane;
  float mv = (v[(long)s_ * 128 + f] + eh[(long)e * 128 + f]) * alpha;
  atomicAdd(&outb[(long)d_ * 128 + f], mv);
}

// ---------------- BatchNorm column stats (64 rows per block, one atomic per block/col)
__global__ __launch_bounds__(128) void k_colstats(const float* __restrict__ Y,
                                                  float* __restrict__ cs, float* __restrict__ cq, int M) {
  int c = threadIdx.x;
  long r0 = (long)blockIdx.x * 64;
  float s = 0.f, q = 0.f;
  for (int j = 0; j < 64; ++j) {
    long r = r0 + j;
    if (r < M) { float v = Y[r * 128 + c]; s += v; q += v * v; }
  }
  atomicAdd(&cs[c], s);
  atomicAdd(&cq[c], q);
}

// ---------------- BN apply + GELU + residual; updates h (f32+bf16) and fused accumulator
__global__ void k_bn_apply(const float* __restrict__ Y, const float* __restrict__ cs,
                           const float* __restrict__ cq, const float* __restrict__ g,
                           const float* __restrict__ b, float* __restrict__ h,
                           bf16_t* __restrict__ hbf, float* __restrict__ hfacc,
                           const float* __restrict__ fw, int layer, int M) {
  int i = blockIdx.x * blockDim.x + threadIdx.x;
  if (i >= M * 128) return;
  int c = i & 127;
  float invN = 1.0f / (float)M;
  float mu = cs[c] * invN;
  float var = cq[c] * invN - mu * mu;
  float xn = (Y[i] - mu) * rsqrtf(var + 1e-5f) * g[c] + b[c];
  float hn = gelu_f(xn) + h[i];
  h[i] = hn;
  hbf[i] = (bf16_t)hn;
  hfacc[i] += fw[layer] * hn;
}

__global__ void k_fusionw(const float* __restrict__ fa, float* __restrict__ fw) {
  if (threadIdx.x == 0 && blockIdx.x == 0) {
    float s[4]; float mx = -1e30f;
    for (int l = 0; l < 4; ++l) {
      float a = 0.f;
      for (int j = 0; j < 128; ++j) a += fa[l * 128 + j];
      s[l] = a * (1.0f / 128.0f);
      mx = fmaxf(mx, s[l]);
    }
    float d = 0.f;
    for (int l = 0; l < 4; ++l) { s[l] = expf(s[l] - mx); d += s[l]; }
    for (int l = 0; l < 4; ++l) fw[l] = s[l] / d;
  }
}

__global__ void k_hf2bf(const float* __restrict__ hf, bf16_t* __restrict__ hfbf, int total) {
  int i = blockIdx.x * blockDim.x + threadIdx.x;
  if (i < total) hfbf[i] = (bf16_t)hf[i];
}

// ---------------- edge classifier head: sigmoid(Z @ w2 + b2)
__global__ __launch_bounds__(256) void k_edge_out(const float* __restrict__ Z,
                                                  const float* __restrict__ w2,
                                                  const float* __restrict__ b2,
                                                  float* __restrict__ out, int E) {
  int w = blockIdx.x * 8 + (threadIdx.x >> 5);
  if (w >= E) return;
  int lane = threadIdx.x & 31;
  float s = 0.f;
  for (int j = lane; j < 128; j += 32) s += Z[(long)w * 128 + j] * w2[j];
  for (int off = 16; off; off >>= 1) s += __shfl_xor(s, off, 32);
  if (lane == 0) out[w] = 1.0f / (1.0f + expf(-(s + b2[0])));
}

// ---------------- node heads: Z @ W2(128x4)+b2 -> mean(2) & softplus(2)+1e-6
__global__ __launch_bounds__(256) void k_head_out(const float* __restrict__ Z,
                                                  const float* __restrict__ w2,
                                                  const float* __restrict__ b2,
                                                  float* __restrict__ outA,
                                                  float* __restrict__ outB, int M) {
  int w = blockIdx.x * 8 + (threadIdx.x >> 5);
  if (w >= M) return;
  int lane = threadIdx.x & 31;
  float s0 = 0.f, s1 = 0.f, s2 = 0.f, s3 = 0.f;
  for (int j = lane; j < 128; j += 32) {
    float z = Z[(long)w * 128 + j];
    s0 += z * w2[j * 4 + 0]; s1 += z * w2[j * 4 + 1];
    s2 += z * w2[j * 4 + 2]; s3 += z * w2[j * 4 + 3];
  }
  for (int off = 16; off; off >>= 1) {
    s0 += __shfl_xor(s0, off, 32); s1 += __shfl_xor(s1, off, 32);
    s2 += __shfl_xor(s2, off, 32); s3 += __shfl_xor(s3, off, 32);
  }
  if (lane == 0) {
    outA[(long)w * 2 + 0] = s0 + b2[0];
    outA[(long)w * 2 + 1] = s1 + b2[1];
    outB[(long)w * 2 + 0] = softplus_f(s2 + b2[2]) + 1e-6f;
    outB[(long)w * 2 + 1] = softplus_f(s3 + b2[3]) + 1e-6f;
  }
}

extern "C" void kernel_launch(void* const* d_in, const int* in_sizes, int n_in,
                              void* d_out, int out_size, void* d_ws, size_t ws_size,
                              hipStream_t stream) {
  (void)n_in; (void)out_size; (void)ws_size;
  const float* x         = (const float*)d_in[0];
  const float* edge_attr = (const float*)d_in[1];
  const float* fourier_B = (const float*)d_in[2];
  const float* nm_w1 = (const float*)d_in[3];  const float* nm_b1 = (const float*)d_in[4];
  const float* nm_g  = (const float*)d_in[5];  const float* nm_be = (const float*)d_in[6];
  const float* nm_w2 = (const float*)d_in[7];  const float* nm_b2 = (const float*)d_in[8];
  const float* ee_w1 = (const float*)d_in[9];  const float* ee_b1 = (const float*)d_in[10];
  const float* ee_g  = (const float*)d_in[11]; const float* ee_be = (const float*)d_in[12];
  const float* ee_w2 = (const float*)d_in[13]; const float* ee_b2 = (const float*)d_in[14];
  const float* q_w = (const float*)d_in[15];   const float* q_b = (const float*)d_in[16];
  const float* k_w = (const float*)d_in[17];   const float* k_b = (const float*)d_in[18];
  const float* v_w = (const float*)d_in[19];   const float* v_b = (const float*)d_in[20];
  const float* ep_w = (const float*)d_in[21];  const float* ep_b = (const float*)d_in[22];
  const float* sk_w = (const float*)d_in[23];  const float* sk_b = (const float*)d_in[24];
  const float* bn_g = (const float*)d_in[25];  const float* bn_b = (const float*)d_in[26];
  const float* fus  = (const float*)d_in[27];
  const float* ec_w1 = (const float*)d_in[28]; const float* ec_b1 = (const float*)d_in[29];
  const float* ec_w2 = (const float*)d_in[30]; const float* ec_b2 = (const float*)d_in[31];
  const float* go_w1 = (const float*)d_in[32]; const float* go_b1 = (const float*)d_in[33];
  const float* go_w2 = (const float*)d_in[34]; const float* go_b2 = (const float*)d_in[35];
  const float* po_w1 = (const float*)d_in[36]; const float* po_b1 = (const float*)d_in[37];
  const float* po_w2 = (const float*)d_in[38]; const float* po_b2 = (const float*)d_in[39];
  const int* eidx = (const int*)d_in[40];

  const int N = in_sizes[0] / 2;
  const int E = in_sizes[1] / 8;
  const int* srcI = eidx;
  const int* dstI = eidx + E;

  float* out_es   = (float*)d_out;        // E
  float* out_go_m = out_es + E;           // N*2
  float* out_go_s = out_go_m + 2 * N;     // N*2
  float* out_po_m = out_go_s + 2 * N;     // N*2
  float* out_po_s = out_po_m + 2 * N;     // N*2

  char* pws = (char*)d_ws;
  auto alloc = [&](size_t bytes) -> void* {
    void* r = (void*)pws;
    pws += (bytes + 255) & ~(size_t)255;
    return r;
  };

  bf16_t* xin  = (bf16_t*)alloc((size_t)N * 96 * 2);
  bf16_t* ein  = (bf16_t*)alloc((size_t)E * 32 * 2);
  float*  big  = (float*) alloc((size_t)E * 128 * 4);  // e1 -> eh -> classifier hidden
  bf16_t* bfE  = (bf16_t*)alloc((size_t)E * 128 * 2);  // edge mid -> e (bf16), persists
  float*  h    = (float*) alloc((size_t)N * 128 * 4);
  bf16_t* bfN  = (bf16_t*)alloc((size_t)N * 128 * 2);  // node mid -> h (bf16)
  float*  qb   = (float*) alloc((size_t)N * 128 * 4);
  float*  kb   = (float*) alloc((size_t)N * 128 * 4);
  float*  vb   = (float*) alloc((size_t)N * 128 * 4);
  float*  outb = (float*) alloc((size_t)N * 128 * 4);
  float*  score= (float*) alloc((size_t)E * 4 * 4);
  unsigned* mkey = (unsigned*)alloc((size_t)N * 4 * 4);
  float*  denom= (float*) alloc((size_t)N * 4 * 4);
  float*  csum = (float*) alloc(128 * 4);
  float*  csq  = (float*) alloc(128 * 4);
  float*  hfacc= (float*) alloc((size_t)N * 128 * 4);
  bf16_t* hfbf = (bf16_t*)alloc((size_t)N * 128 * 2);
  float*  fw   = (float*) alloc(4 * 4);
  bf16_t* nm_w1t = (bf16_t*)alloc(128 * 96 * 2);
  bf16_t* nm_w2t = (bf16_t*)alloc(128 * 128 * 2);
  bf16_t* ee_w1t = (bf16_t*)alloc(128 * 32 * 2);
  bf16_t* ee_w2t = (bf16_t*)alloc(128 * 128 * 2);
  bf16_t* q_wt   = (bf16_t*)alloc(4 * 128 * 128 * 2);
  bf16_t* k_wt   = (bf16_t*)alloc(4 * 128 * 128 * 2);
  bf16_t* v_wt   = (bf16_t*)alloc(4 * 128 * 128 * 2);
  bf16_t* ep_wt  = (bf16_t*)alloc(4 * 128 * 128 * 2);
  bf16_t* sk_wt  = (bf16_t*)alloc(4 * 128 * 128 * 2);
  bf16_t* ec_w1t = (bf16_t*)alloc(3 * 128 * 128 * 2);
  bf16_t* go_w1t = (bf16_t*)alloc(128 * 128 * 2);
  bf16_t* po_w1t = (bf16_t*)alloc(128 * 128 * 2);

  auto pack = [&](const float* W, bf16_t* Wt, int Kr, int Nc, int Kp) {
    int tot = Nc * Kp;
    k_pack<<<(tot + 255) / 256, 256, 0, stream>>>(W, Wt, Kr, Nc, Kp);
  };
  auto gemm = [&](const bf16_t* X, int ldx, const int* idx, const bf16_t* Wt,
                  const float* bias, float* Y, bf16_t* Ybf, int M, int Kp, int flags) {
    dim3 g((M + 127) / 128), b(256);
    if (Kp == 128)      k_gemm<128><<<g, b, 0, stream>>>(X, ldx, idx, Wt, bias, Y, Ybf, M, flags);
    else if (Kp == 96)  k_gemm<96><<<g, b, 0, stream>>>(X, ldx, idx, Wt, bias, Y, Ybf, M, flags);
    else                k_gemm<32><<<g, b, 0, stream>>>(X, ldx, idx, Wt, bias, Y, Ybf, M, flags);
  };

  // ---- weight packing (bf16, transposed, K padded to mult of 32)
  pack(nm_w1, nm_w1t, 66, 128, 96);
  pack(nm_w2, nm_w2t, 128, 128, 128);
  pack(ee_w1, ee_w1t, 8, 128, 32);
  pack(ee_w2, ee_w2t, 128, 128, 128);
  for (int L = 0; L < 4; ++L) {
    pack(q_w + L * 16384, q_wt + L * 16384, 128, 128, 128);
    pack(k_w + L * 16384, k_wt + L * 16384, 128, 128, 128);
    pack(v_w + L * 16384, v_wt + L * 16384, 128, 128, 128);
    pack(ep_w + L * 16384, ep_wt + L * 16384, 128, 128, 128);
    pack(sk_w + L * 16384, sk_wt + L * 16384, 128, 128, 128);
  }
  for (int s = 0; s < 3; ++s) pack(ec_w1 + s * 16384, ec_w1t + s * 16384, 128, 128, 128);
  pack(go_w1, go_w1t, 128, 128, 128);
  pack(po_w1, po_w1t, 128, 128, 128);

  k_fusionw<<<1, 32, 0, stream>>>(fus, fw);
  (void)hipMemsetAsync(hfacc, 0, (size_t)N * 128 * 4, stream);

  // ---- node MLP
  k_build_xin<<<(N * 32 + 255) / 256, 256, 0, stream>>>(x, fourier_B, xin, N);
  gemm(xin, 96, nullptr, nm_w1t, nm_b1, big, nullptr, N, 96, 0);
  k_ln_act<<<N, 128, 0, stream>>>(big, nm_g, nm_be, bfN, N);
  gemm(bfN, 128, nullptr, nm_w2t, nm_b2, h, bfN, N, 128, GF_STOREBF);  // in-place bf16 ok

  // ---- edge MLP
  k_build_ein<<<(E * 32 + 255) / 256, 256, 0, stream>>>(edge_attr, ein, E);
  gemm(ein, 32, nullptr, ee_w1t, ee_b1, big, nullptr, E, 32, 0);
  k_ln_act<<<E, 128, 0, stream>>>(big, ee_g, ee_be, bfE, E);
  gemm(bfE, 128, nullptr, ee_w2t, ee_b2, big, bfE, E, 128, GF_STOREBF); // e kept only in bf16

  // ---- 4 TransformerConv layers
  for (int L = 0; L < 4; ++L) {
    const long WO = (long)L * 16384;
    gemm(bfN, 128, nullptr, q_wt + WO, q_b + L * 128, qb, nullptr, N, 128, 0);
    gemm(bfN, 128, nullptr, k_wt + WO, k_b + L * 128, kb, nullptr, N, 128, 0);
    gemm(bfN, 128, nullptr, v_wt + WO, v_b + L * 128, vb, nullptr, N, 128, 0);
    gemm(bfE, 128, nullptr, ep_wt + WO, ep_b + L * 128, big, nullptr, E, 128, 0); // eh

    (void)hipMemsetAsync(outb, 0, (size_t)N * 128 * 4, stream);
    (void)hipMemsetAsync(mkey, 0, (size_t)N * 4 * 4, stream);
    (void)hipMemsetAsync(denom, 0, (size_t)N * 4 * 4, stream);

    int EH = E * 4;
    k_score_max<<<(EH + 7) / 8, 256, 0, stream>>>(qb, kb, big, srcI, dstI, score, mkey, E);
    k_exp_denom<<<(EH + 255) / 256, 256, 0, stream>>>(score, mkey, denom, dstI, EH);
    k_msg<<<(EH + 7) / 8, 256, 0, stream>>>(vb, big, score, denom, srcI, dstI, outb, E);

    gemm(bfN, 128, nullptr, sk_wt + WO, sk_b + L * 128, outb, nullptr, N, 128, GF_ACCUM); // + skip

    (void)hipMemsetAsync(csum, 0, 128 * 4, stream);
    (void)hipMemsetAsync(csq, 0, 128 * 4, stream);
    k_colstats<<<(N + 63) / 64, 128, 0, stream>>>(outb, csum, csq, N);
    k_bn_apply<<<(N * 128 + 255) / 256, 256, 0, stream>>>(
        outb, csum, csq, bn_g + L * 128, bn_b + L * 128, h, bfN, hfacc, fw, L, N);
  }

  // ---- fusion + heads
  k_hf2bf<<<(N * 128 + 255) / 256, 256, 0, stream>>>(hfacc, hfbf, N * 128);

  gemm(hfbf, 128, srcI, ec_w1t,            nullptr, big, nullptr, E, 128, 0);
  gemm(hfbf, 128, dstI, ec_w1t + 16384,    nullptr, big, nullptr, E, 128, GF_ACCUM);
  gemm(bfE,  128, nullptr, ec_w1t + 32768, ec_b1,   big, nullptr, E, 128, GF_ACCUM | GF_GELU);
  k_edge_out<<<(E + 7) / 8, 256, 0, stream>>>(big, ec_w2, ec_b2, out_es, E);

  gemm(hfbf, 128, nullptr, go_w1t, go_b1, qb, nullptr, N, 128, GF_GELU);
  k_head_out<<<(N + 7) / 8, 256, 0, stream>>>(qb, go_w2, go_b2, out_go_m, out_go_s, N);

  gemm(hfbf, 128, nullptr, po_w1t, po_b1, kb, nullptr, N, 128, GF_GELU);
  k_head_out<<<(N + 7) / 8, 256, 0, stream>>>(kb, po_w2, po_b2, out_po_m, out_po_s, N);
}
